// Decoder_46420006535519
// MI455X (gfx1250) — matrix operand order repaired and checked by reading, not compiled, not run on hardware
//
#include <hip/hip_runtime.h>
#include <math.h>

typedef _Float16 h16v __attribute__((ext_vector_type(16)));
typedef _Float16 h8v  __attribute__((ext_vector_type(8)));
typedef _Float16 h4v  __attribute__((ext_vector_type(4)));
typedef float    f8v  __attribute__((ext_vector_type(8)));
typedef float    f4v  __attribute__((ext_vector_type(4)));

// ---- problem constants (from reference) ----
#define Bc   4
#define MHCc 200
#define PEPc 56
#define Nc   256
#define Sc   384
#define Zc   384
#define Hc   8
#define DKc  16
#define DVc  16
#define PQc  4
#define PVc  4
#define RDc  128
#define NBc  8
#define NRc  2
#define NAc  7
#define OCc  3328          // H*(DV+Z+PV*4)
#define EPSc 1e-8f

// output layout offsets (floats)
#define OFF_UN 393216      // B*N*S
#define OFF_AN 396352      // +B*PEP*NA*2
#define OFF_R  399488      // +B*PEP*NA*2
#define OFF_T  401504      // +B*PEP*9

// =====================================================================
// Batched WMMA GEMM:  C = act( (reluA?relu(A)) @ B + bias + res )
// A: [M,K] fp32 or f16 ;  B: [K,>=N] f16, columns padded to bNfull (mult 64)
// batch offset for X = (bz / xMod)*xSo + (bz % xMod)*xSi
// LDS tiles staged in WMMA-fragment-ready layout:
//   A 16x32 frag:  lane = m + 16*((k>>3)&1),  elem = (k&7) + 8*(k>>4)
//   B 32x16 frag:  lane = (n&15) + 16*((k>>3)&1), elem = (k&7) + 8*(k>>4)
// A rows are clamped+zero-selected so partial-M tiles stay vectorized.
// =====================================================================
struct GemmArgs {
  const void* A; const _Float16* Bm;
  const float* bias; const float* res; float* C;
  long aMod, aSo, aSi, bMod, bSo, bSi, cMod, cSo, cSi;
  int lda, ldb, ldc, M, N, K, bNfull;
  int reluOut;
};

template<int AF16, int RELUA>
__global__ void __launch_bounds__(256) gemm_wmma_k(GemmArgs g) {
  __shared__ __align__(32) _Float16 AsF[4][32][16];   // [wm slot][lane][elem]
  __shared__ __align__(32) _Float16 BsF[4][32][16];   // [n subtile][lane][elem]
  const int tid  = threadIdx.x;
  const int lane = tid & 31;
  const int wave = tid >> 5;
  const int wm   = wave & 3, wn = wave >> 2;
  const int m0 = blockIdx.y * 64, n0 = blockIdx.x * 64;
  const int bz = blockIdx.z;
  const long aOff = (long)(bz / g.aMod) * g.aSo + (long)(bz % g.aMod) * g.aSi;
  const long bOff = (long)(bz / g.bMod) * g.bSo + (long)(bz % g.bMod) * g.bSi;
  const long cOff = (long)(bz / g.cMod) * g.cSo + (long)(bz % g.cMod) * g.cSi;
  const float*    __restrict__ Af = (const float*)g.A;
  const _Float16* __restrict__ Ah = (const _Float16*)g.A;
  const _Float16* __restrict__ Bh = g.Bm;

  const bool kAligned = (g.K & 31) == 0;
  const bool fullB = kAligned && (n0 + 64 <= g.bNfull);

  // per-thread staging geometry (fast paths)
  const int ar = tid >> 2;          // A row in tile (0..63)
  const int aq = tid & 3;           // A 8-wide k chunk
  const int aLane = (ar & 15) + ((aq & 1) << 4);
  const int aSlot = ar >> 4;
  const int aEoff = (aq >> 1) << 3;
  const bool aOk  = true;           // row validity handled per-iteration
  const int bnp = tid & 15;         // B 4-wide n chunk
  const int bkp = tid >> 4;         // B k pair (0..15)
  const int bE  = ((bkp << 1) & 7) + ((bkp >> 3) << 3);
  const int bLaneBase = ((bkp >> 2) & 1) << 4;

  // clamped A row (always a valid address; OOB rows zero-selected)
  const int growA = m0 + ar;
  const bool okA  = growA < g.M;
  const int crowA = okA ? growA : (g.M - 1);

  f8v acc0 = {0.f,0.f,0.f,0.f,0.f,0.f,0.f,0.f};
  f8v acc1 = {0.f,0.f,0.f,0.f,0.f,0.f,0.f,0.f};

  for (int k0 = 0; k0 < g.K; k0 += 32) {
    // ---------------- stage A (vectorized, row-clamped) ----------------
    if (kAligned) {
      long base = aOff + (long)crowA * g.lda + k0 + aq*8;
      h8v h;
      if (AF16) {
        h = *(const h8v*)(Ah + base);
        if (RELUA) {
          #pragma unroll
          for (int j = 0; j < 8; ++j) h[j] = (h[j] > (_Float16)0) ? h[j] : (_Float16)0;
        }
      } else {
        f4v v0 = *(const f4v*)(Af + base);
        f4v v1 = *(const f4v*)(Af + base + 4);
        #pragma unroll
        for (int j = 0; j < 4; ++j) {
          float a = v0[j], b = v1[j];
          if (RELUA) { a = fmaxf(a, 0.f); b = fmaxf(b, 0.f); }
          h[j]   = (_Float16)a;
          h[j+4] = (_Float16)b;
        }
      }
      #pragma unroll
      for (int j = 0; j < 8; ++j) h[j] = okA ? h[j] : (_Float16)0;
      *(h8v*)&AsF[aSlot][aLane][aEoff] = h;
    } else {
      #pragma unroll
      for (int i = 0; i < 8; ++i) {
        int lin = tid + i*256;
        int r = lin >> 5, c = lin & 31;
        int gr = m0 + r, gc = k0 + c;
        bool inb = (gr < g.M) && (gc < g.K);
        long idx = inb ? (aOff + (long)gr * g.lda + gc) : aOff;
        float v = AF16 ? (float)Ah[idx] : Af[idx];
        v = inb ? v : 0.f;
        if (RELUA) v = fmaxf(v, 0.f);
        AsF[r >> 4][(r & 15) + (((c >> 3) & 1) << 4)][(c & 7) + ((c >> 4) << 3)] = (_Float16)v;
      }
    }
    // ---------------- stage B (f16, padded -> always vectorized) -------
    if (fullB) {
      long base = bOff + (long)(k0 + bkp*2) * g.ldb + n0 + bnp*4;
      h4v r0 = *(const h4v*)(Bh + base);
      h4v r1 = *(const h4v*)(Bh + base + g.ldb);
      #pragma unroll
      for (int j = 0; j < 4; ++j) {
        union { _Float16 h[2]; unsigned u; } pk;
        pk.h[0] = r0[j]; pk.h[1] = r1[j];
        int n = bnp*4 + j;
        *(unsigned*)&BsF[n >> 4][(n & 15) + bLaneBase][bE] = pk.u;
      }
    } else {
      #pragma unroll
      for (int i = 0; i < 8; ++i) {
        int lin = tid + i*256;
        int r = lin >> 6, c = lin & 63;      // r = k (0..31), c = n (0..63)
        int gr = k0 + r, gc = n0 + c;
        bool inb = (gr < g.K) && (gc < g.bNfull);
        long idx = inb ? (bOff + (long)gr * g.ldb + gc) : bOff;
        float v = (float)Bh[idx];
        v = inb ? v : 0.f;
        BsF[c >> 4][(c & 15) + (((r >> 3) & 1) << 4)][(r & 7) + ((r >> 4) << 3)] = (_Float16)v;
      }
    }
    __syncthreads();

    h16v af  = *(const h16v*)&AsF[wm][lane][0];
    h16v bf0 = *(const h16v*)&BsF[wn*2 + 0][lane][0];
    h16v bf1 = *(const h16v*)&BsF[wn*2 + 1][lane][0];
    acc0 = __builtin_amdgcn_wmma_f32_16x16x32_f16(false, af, false, bf0, (short)0, acc0, false, false);
    acc1 = __builtin_amdgcn_wmma_f32_16x16x32_f16(false, af, false, bf1, (short)0, acc1, false, false);
    __syncthreads();
  }

  // epilogue: C layout -> VGPR v: M = v + 8*(lane>=16), N = lane&15
  #pragma unroll
  for (int v = 0; v < 8; ++v) {
    int ml = v + ((lane >> 4) << 3);
    int gr = m0 + wm*16 + ml;
    if (gr >= g.M) continue;
    int gc0 = n0 + wn*32 + (lane & 15);
    int gc1 = gc0 + 16;
    if (gc0 < g.N) {
      float x = acc0[v];
      if (g.bias) x += g.bias[gc0];
      if (g.res)  x += g.res[cOff + (long)gr*g.ldc + gc0];
      if (g.reluOut) x = fmaxf(x, 0.f);
      g.C[cOff + (long)gr*g.ldc + gc0] = x;
    }
    if (gc1 < g.N) {
      float x = acc1[v];
      if (g.bias) x += g.bias[gc1];
      if (g.res)  x += g.res[cOff + (long)gr*g.ldc + gc1];
      if (g.reluOut) x = fmaxf(x, 0.f);
      g.C[cOff + (long)gr*g.ldc + gc1] = x;
    }
  }
}

// weight pre-pack: fp32 [K,N] -> f16 [K,ldo] zero-padded columns
__global__ void pack_w16_k(const float* __restrict__ w, _Float16* __restrict__ o,
                           int K, int N, int ldo) {
  long idx = (long)blockIdx.x*256 + threadIdx.x;
  if (idx >= (long)K*ldo) return;
  int r = (int)(idx / ldo), c = (int)(idx % ldo);
  o[idx] = (_Float16)((c < N) ? w[(long)r*N + c] : 0.f);
}

// =====================================================================
// LayerNorm over last dim (optional residual add; fp32 and/or f16 out)
// =====================================================================
__global__ void __launch_bounds__(128) ln_rows_k(const float* __restrict__ x,
                                                 const float* __restrict__ res,
                                                 const float* __restrict__ g,
                                                 const float* __restrict__ b,
                                                 int D, float* __restrict__ outF,
                                                 _Float16* __restrict__ outH) {
  const long row = blockIdx.x;
  const int tid = threadIdx.x;
  const long base = row * (long)D;
  float vals[4];
  int cnt = 0;
  float s = 0.f, ss = 0.f;
  for (int idx = tid; idx < D; idx += 128) {
    float v = x[base + idx];
    if (res) v += res[base + idx];
    vals[cnt++] = v;
    s += v; ss += v*v;
  }
  __shared__ float r1[128], r2[128];
  r1[tid] = s; r2[tid] = ss;
  __syncthreads();
  for (int off = 64; off > 0; off >>= 1) {
    if (tid < off) { r1[tid] += r1[tid+off]; r2[tid] += r2[tid+off]; }
    __syncthreads();
  }
  float mean = r1[0] / (float)D;
  float var  = r2[0] / (float)D - mean*mean;
  float rstd = rsqrtf(var + 1e-5f);
  cnt = 0;
  for (int idx = tid; idx < D; idx += 128) {
    float v = (vals[cnt++] - mean) * rstd * g[idx] + b[idx];
    if (outF) outF[base + idx] = v;
    if (outH) outH[base + idx] = (_Float16)v;
  }
}

// =====================================================================
// Row softmax, D == 256, block 256, in-place
// =====================================================================
__global__ void __launch_bounds__(256) softmax_rows_k(float* __restrict__ x) {
  const long row = blockIdx.x;
  float* p = x + row * (long)Nc;
  const int tid = threadIdx.x;
  float v = p[tid];
  __shared__ float sm[256];
  sm[tid] = v; __syncthreads();
  for (int off = 128; off > 0; off >>= 1) { if (tid < off) sm[tid] = fmaxf(sm[tid], sm[tid+off]); __syncthreads(); }
  float mx = sm[0]; __syncthreads();
  float e = __expf(v - mx);
  sm[tid] = e; __syncthreads();
  for (int off = 128; off > 0; off >>= 1) { if (tid < off) sm[tid] += sm[tid+off]; __syncthreads(); }
  p[tid] = e / sm[0];
}

// =====================================================================
// frames
// =====================================================================
__global__ void init_frames_k(float* pepR, float* pepT) {
  int idx = blockIdx.x*256 + threadIdx.x;
  if (idx >= Bc*PEPc) return;
  float* R = pepR + idx*9;
  R[0]=1.f; R[1]=0.f; R[2]=0.f; R[3]=0.f; R[4]=1.f; R[5]=0.f; R[6]=0.f; R[7]=0.f; R[8]=1.f;
  float* t = pepT + idx*3;
  t[0]=0.f; t[1]=0.f; t[2]=0.f;
}

__global__ void build_frames_k(const float* __restrict__ mhcbb, const float* __restrict__ pepR,
                               const float* __restrict__ pepT,
                               float* __restrict__ Rbuf, float* __restrict__ tbuf) {
  int idx = blockIdx.x*256 + threadIdx.x;
  if (idx >= Bc*Nc) return;
  int n = idx % Nc, b = idx / Nc;
  float* R = Rbuf + (long)idx*9;
  float* t = tbuf + (long)idx*3;
  if (n < MHCc) {
    const float* src = mhcbb + ((long)(b*MHCc + n))*16;
    #pragma unroll
    for (int i = 0; i < 3; ++i) {
      R[i*3+0]=src[i*4+0]; R[i*3+1]=src[i*4+1]; R[i*3+2]=src[i*4+2];
      t[i]=src[i*4+3];
    }
  } else {
    int np = n - MHCc;
    const float* sr = pepR + (long)(b*PEPc + np)*9;
    const float* st = pepT + (long)(b*PEPc + np)*3;
    #pragma unroll
    for (int i = 0; i < 9; ++i) R[i] = sr[i];
    t[0]=st[0]; t[1]=st[1]; t[2]=st[2];
  }
}

// xg_r = sum_c x_c * R[c,r] + t_r
__global__ void to_global_k(const float* __restrict__ xp, const float* __restrict__ Rb,
                            const float* __restrict__ tb, float* __restrict__ xg, int npts) {
  long idx = (long)blockIdx.x*256 + threadIdx.x;
  long total = (long)Bc*Nc*npts;
  if (idx >= total) return;
  long bn = idx / npts;
  const float* R = Rb + bn*9;
  const float* t = tb + bn*3;
  long base = idx*3;
  float x0 = xp[base], x1 = xp[base+1], x2 = xp[base+2];
  #pragma unroll
  for (int r = 0; r < 3; ++r)
    xg[base + r] = x0*R[0*3+r] + x1*R[1*3+r] + x2*R[2*3+r] + t[r];
}

__global__ void sqsum_k(const float* __restrict__ xg, float* __restrict__ out) {
  int idx = blockIdx.x*256 + threadIdx.x;
  if (idx >= Bc*Nc*Hc) return;
  const float* p = xg + (long)idx*(PQc*3);
  float s = 0.f;
  #pragma unroll
  for (int d = 0; d < PQc*3; ++d) s += p[d]*p[d];
  out[idx] = s;
}

// =====================================================================
// attention logits: block per (b,h,i), thread per j
// =====================================================================
__global__ void __launch_bounds__(256) logits_k(const float* __restrict__ q,
                                                const float* __restrict__ k,
                                                const float* __restrict__ qg,
                                                const float* __restrict__ kg,
                                                const float* __restrict__ qsq,
                                                const float* __restrict__ ksq,
                                                const float* __restrict__ zb,
                                                const float* __restrict__ pwraw,
                                                float* __restrict__ logits) {
  int idx = blockIdx.x;
  int i = idx % Nc, h = (idx / Nc) % Hc, b = idx / (Nc*Hc);
  int tid = threadIdx.x;
  __shared__ float qi[DKc], qgi[PQc*3], sh[2];
  if (tid < DKc) qi[tid] = q[((long)(b*Nc + i))*(Hc*DKc) + h*DKc + tid];
  if (tid >= 32 && tid < 32 + PQc*3)
    qgi[tid-32] = qg[((long)(b*Nc + i)*Hc + h)*(PQc*3) + (tid-32)];
  if (tid == 64) sh[0] = qsq[(b*Nc + i)*Hc + h];
  if (tid == 65) { float w = pwraw[h]; sh[1] = (w > 20.f) ? w : log1pf(__expf(w)); }
  __syncthreads();
  int j = tid;
  const float* kr = k + ((long)(b*Nc + j))*(Hc*DKc) + h*DKc;
  float dot = 0.f;
  #pragma unroll
  for (int d = 0; d < DKc; ++d) dot += qi[d]*kr[d];
  const float* kgr = kg + ((long)(b*Nc + j)*Hc + h)*(PQc*3);
  float cross = 0.f;
  #pragma unroll
  for (int d = 0; d < PQc*3; ++d) cross += qgi[d]*kgr[d];
  float dist = sh[0] + ksq[(b*Nc + j)*Hc + h] - 2.f*cross;
  float zbv  = zb[(((long)(b*Nc + i))*Nc + j)*Hc + h];
  float lg = 0.14433757f*dot + 0.57735027f*zbv - 0.5f*0.13608276f*sh[1]*dist;
  logits[(((long)(b*Hc + h))*Nc + i)*Nc + j] = lg;
}

// o_s[b,i,h,d] = sum_j attn * v  -> o_concat[.. h*16+d]
__global__ void attn_os_k(const float* __restrict__ attn, const float* __restrict__ v,
                          float* __restrict__ oc) {
  long idx = (long)blockIdx.x*256 + threadIdx.x;
  if (idx >= (long)Bc*Nc*Hc*DVc) return;
  int d = idx & 15, h = (int)((idx >> 4) & 7);
  int i = (int)((idx >> 7) % Nc), b = (int)(idx / ((long)128*Nc));
  const float* ar = attn + (((long)(b*Hc + h))*Nc + i)*Nc;
  const float* vb = v + (long)b*Nc*Hc*DVc + h*DVc + d;
  float s = 0.f;
  for (int j = 0; j < Nc; ++j) s += ar[j] * vb[(long)j*Hc*DVc];
  oc[((long)(b*Nc + i))*OCc + h*DVc + d] = s;
}

// og[b,i,h,p,c] = sum_j attn * vg
__global__ void attn_og_k(const float* __restrict__ attn, const float* __restrict__ vg,
                          float* __restrict__ og) {
  long idx = (long)blockIdx.x*256 + threadIdx.x;
  if (idx >= (long)Bc*Nc*Hc*PVc*3) return;
  int c = (int)(idx % 3);
  int p = (int)((idx/3) & 3);
  int h = (int)((idx/12) & 7);
  int i = (int)((idx/96) % Nc);
  int b = (int)(idx / ((long)96*Nc));
  const float* ar  = attn + (((long)(b*Hc + h))*Nc + i)*Nc;
  const float* vgb = vg + (long)b*Nc*(Hc*PVc*3) + h*(PVc*3) + p*3 + c;
  float s = 0.f;
  for (int j = 0; j < Nc; ++j) s += ar[j] * vgb[(long)j*(Hc*PVc*3)];
  og[idx] = s;
}

// ol_c = sum_r (og_r - t_r) * R[c,r]; write o_p and o_n into o_concat
__global__ void finalize_pts_k(const float* __restrict__ og, const float* __restrict__ Rb,
                               const float* __restrict__ tb, float* __restrict__ oc) {
  long idx = (long)blockIdx.x*256 + threadIdx.x;
  if (idx >= (long)Bc*Nc*Hc*PVc) return;
  int p = (int)(idx & 3);
  int h = (int)((idx >> 2) & 7);
  long bn = idx >> 5;                 // b*Nc + n
  const float* R = Rb + bn*9;
  const float* t = tb + bn*3;
  const float* o = og + idx*3;
  float d0 = o[0]-t[0], d1 = o[1]-t[1], d2 = o[2]-t[2];
  float* row = oc + bn*OCc;
  float nsum = 0.f;
  #pragma unroll
  for (int c = 0; c < 3; ++c) {
    float ol = d0*R[c*3+0] + d1*R[c*3+1] + d2*R[c*3+2];
    row[Hc*DVc + Hc*Zc + h*(PVc*3) + p*3 + c] = ol;
    nsum += ol*ol;
  }
  row[Hc*DVc + Hc*Zc + Hc*PVc*3 + h*PVc + p] = sqrtf(nsum + EPSc);
}

// quaternion/backbone update
__global__ void quat_update_k(const float* __restrict__ upd, float* pepR, float* pepT) {
  int idx = blockIdx.x*256 + threadIdx.x;
  if (idx >= Bc*PEPc) return;
  const float* u = upd + (long)idx*6;
  float x=u[0], y=u[1], z=u[2], tx=u[3], ty=u[4], tz=u[5];
  float inv = rsqrtf(1.f + x*x + y*y + z*z);
  float w = inv; x*=inv; y*=inv; z*=inv;
  float Ru[9] = {1.f-2.f*(y*y+z*z), 2.f*(x*y-w*z),     2.f*(x*z+w*y),
                 2.f*(x*y+w*z),     1.f-2.f*(x*x+z*z), 2.f*(y*z-w*x),
                 2.f*(x*z-w*y),     2.f*(y*z+w*x),     1.f-2.f*(x*x+y*y)};
  float* Rp = pepR + (long)idx*9;
  float* tp = pepT + (long)idx*3;
  float nt[3], nR[9];
  #pragma unroll
  for (int r = 0; r < 3; ++r)
    nt[r] = Rp[r*3+0]*tx + Rp[r*3+1]*ty + Rp[r*3+2]*tz + tp[r];
  #pragma unroll
  for (int r = 0; r < 3; ++r)
    #pragma unroll
    for (int d = 0; d < 3; ++d)
      nR[r*3+d] = Rp[r*3+0]*Ru[0*3+d] + Rp[r*3+1]*Ru[1*3+d] + Rp[r*3+2]*Ru[2*3+d];
  #pragma unroll
  for (int i = 0; i < 9; ++i) Rp[i] = nR[i];
  tp[0]=nt[0]; tp[1]=nt[1]; tp[2]=nt[2];
}

// angle head: write unnorm + normalized angles straight to d_out
__global__ void angles_out_k(const float* __restrict__ o, float* __restrict__ dout) {
  int idx = blockIdx.x*256 + threadIdx.x;
  if (idx >= Bc*PEPc*NAc) return;
  int na = idx % NAc, r = idx / NAc;
  float x0 = o[(long)r*(NAc*2) + na*2 + 0];
  float x1 = o[(long)r*(NAc*2) + na*2 + 1];
  float dn = sqrtf(fmaxf(x0*x0 + x1*x1, EPSc));
  dout[OFF_UN + idx*2 + 0] = x0;
  dout[OFF_UN + idx*2 + 1] = x1;
  dout[OFF_AN + idx*2 + 0] = x0/dn;
  dout[OFF_AN + idx*2 + 1] = x1/dn;
}

__global__ void pack_tail_k(const float* __restrict__ pepR, const float* __restrict__ pepT,
                            float* __restrict__ dout) {
  int idx = blockIdx.x*256 + threadIdx.x;
  if (idx < Bc*PEPc*9) dout[OFF_R + idx] = pepR[idx];
  if (idx < Bc*PEPc*3) dout[OFF_T + idx] = pepT[idx] * 10.f;
}

// =====================================================================
// host side
// =====================================================================
struct PW { const _Float16* p; int ld; };   // packed f16 weight, ld = ceil64(N)

static void gemm_launch(hipStream_t st,
    const void* A, int aF16, long aMod, long aSo, long aSi, int lda,
    const _Float16* Bm, long bMod, long bSo, long bSi, int ldb, int bNfull,
    const float* bias, const float* res,
    float* C, long cMod, long cSo, long cSi, int ldc,
    int M, int N, int K, int batch, int reluA, int reluOut) {
  GemmArgs g;
  g.A = A; g.Bm = Bm; g.bias = bias; g.res = res; g.C = C;
  g.aMod = aMod ? aMod : 1; g.aSo = aSo; g.aSi = aSi;
  g.bMod = bMod ? bMod : 1; g.bSo = bSo; g.bSi = bSi;
  g.cMod = cMod ? cMod : 1; g.cSo = cSo; g.cSi = cSi;
  g.lda = lda; g.ldb = ldb; g.ldc = ldc; g.M = M; g.N = N; g.K = K;
  g.bNfull = bNfull; g.reluOut = reluOut;
  dim3 grid((N + 63)/64, (M + 63)/64, batch);
  if (aF16)       gemm_wmma_k<1,0><<<grid, dim3(256), 0, st>>>(g);
  else if (reluA) gemm_wmma_k<0,1><<<grid, dim3(256), 0, st>>>(g);
  else            gemm_wmma_k<0,0><<<grid, dim3(256), 0, st>>>(g);
}

static void gemm_plain(hipStream_t st, const float* A, int lda, PW W,
                       const float* bias, const float* res, float* C, int ldc,
                       int M, int N, int K, int reluA, int reluOut) {
  gemm_launch(st, A, 0, 1, 0, 0, lda, W.p, 1, 0, 0, W.ld, W.ld, bias, res,
              C, 1, 0, 0, ldc, M, N, K, 1, reluA, reluOut);
}

extern "C" void kernel_launch(void* const* d_in, const int* in_sizes, int n_in,
                              void* d_out, int out_size, void* d_ws, size_t ws_size,
                              hipStream_t stream) {
  (void)in_sizes; (void)n_in; (void)out_size; (void)ws_size;
  const float* single = (const float*)d_in[0];
  const float* pair   = (const float*)d_in[1];
  /* d_in[2] = aatype: unused in forward */
  const float* mhcbb  = (const float*)d_in[3];
  int ix = 4;
  #define FP() ((const float*)d_in[ix++])
  const float *ln_s_g=FP(), *ln_s_b=FP(), *ln_z_g=FP(), *ln_z_b=FP();
  const float *lin_in_w=FP(), *lin_in_b=FP();
  const float *wq=FP(), *wk=FP(), *wv=FP();
  const float *wpq=FP(), *wpk=FP(), *wpv=FP();
  const float *pw=FP();
  const float *wpb=FP(), *bpb=FP();
  const float *wo=FP(), *bo=FP();
  const float *ln_attn_g=FP(), *ln_attn_b=FP();
  const float *ff_w1=FP(), *ff_b1=FP(), *ff_w2=FP(), *ff_b2=FP(), *ff_w3=FP(), *ff_b3=FP();
  const float *ln_ff_g=FP(), *ln_ff_b=FP();
  const float *ln_ipa_g=FP(), *ln_ipa_b=FP();
  const float *tw1=FP(), *tb1=FP(), *tw2=FP(), *tb2=FP(), *tw3=FP(), *tb3=FP();
  const float *ln_t_g=FP(), *ln_t_b=FP();
  const float *bb_w=FP(), *bb_b=FP();
  const float *ar_in_w=FP(), *ar_in_b=FP(), *ar_init_w=FP(), *ar_init_b=FP();
  const float *arw[NRc][4];
  for (int r = 0; r < NRc; ++r) { arw[r][0]=FP(); arw[r][1]=FP(); arw[r][2]=FP(); arw[r][3]=FP(); }
  const float *ar_out_w=FP(), *ar_out_b=FP();
  #undef FP

  // ---- workspace carve ----
  char* wsb = (char*)d_ws;
  size_t off = 0;
  auto alloc = [&](size_t bytes) -> void* {
    void* p = wsb + off; off += (bytes + 255) & ~(size_t)255; return p;
  };
  _Float16* z16 = (_Float16*)alloc((size_t)Bc*Nc*Nc*Zc*sizeof(_Float16));   // 201 MB
  float* zb     = (float*)alloc((size_t)Bc*Nc*Nc*Hc*4);                     // (b,i,j,h)
  float* sInit  = (float*)alloc((size_t)Bc*Nc*Sc*4);
  float* sBuf   = (float*)alloc((size_t)Bc*Nc*Sc*4);
  float* t0     = (float*)alloc((size_t)Bc*Nc*Sc*4);
  float* t1     = (float*)alloc((size_t)Bc*Nc*Sc*4);
  float* t2     = (float*)alloc((size_t)Bc*Nc*Sc*4);
  float* qB     = (float*)alloc((size_t)Bc*Nc*Hc*DKc*4);
  float* kB     = (float*)alloc((size_t)Bc*Nc*Hc*DKc*4);
  float* vB     = (float*)alloc((size_t)Bc*Nc*Hc*DVc*4);
  float* qpB    = (float*)alloc((size_t)Bc*Nc*Hc*PQc*3*4);
  float* kpB    = (float*)alloc((size_t)Bc*Nc*Hc*PQc*3*4);
  float* vpB    = (float*)alloc((size_t)Bc*Nc*Hc*PVc*3*4);
  float* qgB    = (float*)alloc((size_t)Bc*Nc*Hc*PQc*3*4);
  float* kgB    = (float*)alloc((size_t)Bc*Nc*Hc*PQc*3*4);
  float* vgB    = (float*)alloc((size_t)Bc*Nc*Hc*PVc*3*4);
  float* ogB    = (float*)alloc((size_t)Bc*Nc*Hc*PVc*3*4);
  float* qsqB   = (float*)alloc((size_t)Bc*Nc*Hc*4);
  float* ksqB   = (float*)alloc((size_t)Bc*Nc*Hc*4);
  float* attnB  = (float*)alloc((size_t)Bc*Hc*Nc*Nc*4);
  float* ocB    = (float*)alloc((size_t)Bc*Nc*OCc*4);
  float* Rbuf   = (float*)alloc((size_t)Bc*Nc*9*4);
  float* tbuf   = (float*)alloc((size_t)Bc*Nc*3*4);
  float* pepR   = (float*)alloc((size_t)Bc*PEPc*9*4);
  float* pepT   = (float*)alloc((size_t)Bc*PEPc*3*4);
  float* updB   = (float*)alloc((size_t)Bc*PEPc*6*4);
  float* aInit  = (float*)alloc((size_t)Bc*PEPc*RDc*4);
  float* xAr    = (float*)alloc((size_t)Bc*PEPc*RDc*4);
  float* hAr    = (float*)alloc((size_t)Bc*PEPc*RDc*4);
  float* oAr    = (float*)alloc((size_t)Bc*PEPc*NAc*2*4);

  // ---- pre-pack all weights to f16, columns padded to multiple of 64 ----
  auto packW = [&](const float* w, int K, int N) -> PW {
    int ld = (N + 63) & ~63;
    _Float16* p = (_Float16*)alloc((size_t)K*ld*sizeof(_Float16));
    long tot = (long)K*ld;
    pack_w16_k<<<(int)((tot + 255)/256), 256, 0, stream>>>(w, p, K, N, ld);
    return PW{p, ld};
  };
  PW Plin  = packW(lin_in_w, Sc, Sc);
  PW Pwq   = packW(wq,  Sc, Hc*DKc);
  PW Pwk   = packW(wk,  Sc, Hc*DKc);
  PW Pwv   = packW(wv,  Sc, Hc*DVc);
  PW Pwpq  = packW(wpq, Sc, Hc*PQc*3);
  PW Pwpk  = packW(wpk, Sc, Hc*PQc*3);
  PW Pwpv  = packW(wpv, Sc, Hc*PVc*3);
  PW Pwpb  = packW(wpb, Zc, Hc);
  PW Pwo   = packW(wo,  OCc, Sc);
  PW Pff1  = packW(ff_w1, Sc, Sc);
  PW Pff2  = packW(ff_w2, Sc, Sc);
  PW Pff3  = packW(ff_w3, Sc, Sc);
  PW Ptw1  = packW(tw1, Sc, Sc);
  PW Ptw2  = packW(tw2, Sc, Sc);
  PW Ptw3  = packW(tw3, Sc, Sc);
  PW Pbb   = packW(bb_w, Sc, 6);
  PW ParIn = packW(ar_in_w, Sc, RDc);
  PW ParI0 = packW(ar_init_w, Sc, RDc);
  PW Par[NRc][2];
  for (int r = 0; r < NRc; ++r) {
    Par[r][0] = packW(arw[r][0], RDc, RDc);
    Par[r][1] = packW(arw[r][2], RDc, RDc);
  }
  PW ParO  = packW(ar_out_w, RDc, NAc*2);

  float* dout = (float*)d_out;

  // ---- pre-loop ----
  ln_rows_k<<<Bc*Nc, 128, 0, stream>>>(single, nullptr, ln_s_g, ln_s_b, Sc, sInit, nullptr);
  ln_rows_k<<<Bc*Nc*Nc, 128, 0, stream>>>(pair, nullptr, ln_z_g, ln_z_b, Zc, nullptr, z16);
  gemm_plain(stream, sInit, Sc, Plin, lin_in_b, nullptr, sBuf, Sc, Bc*Nc, Sc, Sc, 0, 0);
  // z_bias = z @ wpb + bpb : A is f16, M = B*N*N, N = H (padded B)
  gemm_launch(stream, z16, 1, 1, 0, 0, Zc, Pwpb.p, 1, 0, 0, Pwpb.ld, Pwpb.ld, bpb, nullptr,
              zb, 1, 0, 0, Hc, Bc*Nc*Nc, Hc, Zc, 1, 0, 0);
  init_frames_k<<<(Bc*PEPc + 255)/256, 256, 0, stream>>>(pepR, pepT);
  gemm_launch(stream, sInit + (long)MHCc*Sc, 0, 1, (long)Nc*Sc, 0, Sc,
              ParI0.p, 1, 0, 0, ParI0.ld, ParI0.ld, ar_init_b, nullptr,
              aInit, 1, (long)PEPc*RDc, 0, RDc, PEPc, RDc, Sc, Bc, 1, 0);

  // ---- iterations ----
  for (int it = 0; it < NBc; ++it) {
    build_frames_k<<<(Bc*Nc + 255)/256, 256, 0, stream>>>(mhcbb, pepR, pepT, Rbuf, tbuf);

    gemm_plain(stream, sBuf, Sc, Pwq,  nullptr, nullptr, qB,  Hc*DKc,   Bc*Nc, Hc*DKc,   Sc, 0, 0);
    gemm_plain(stream, sBuf, Sc, Pwk,  nullptr, nullptr, kB,  Hc*DKc,   Bc*Nc, Hc*DKc,   Sc, 0, 0);
    gemm_plain(stream, sBuf, Sc, Pwv,  nullptr, nullptr, vB,  Hc*DVc,   Bc*Nc, Hc*DVc,   Sc, 0, 0);
    gemm_plain(stream, sBuf, Sc, Pwpq, nullptr, nullptr, qpB, Hc*PQc*3, Bc*Nc, Hc*PQc*3, Sc, 0, 0);
    gemm_plain(stream, sBuf, Sc, Pwpk, nullptr, nullptr, kpB, Hc*PQc*3, Bc*Nc, Hc*PQc*3, Sc, 0, 0);
    gemm_plain(stream, sBuf, Sc, Pwpv, nullptr, nullptr, vpB, Hc*PVc*3, Bc*Nc, Hc*PVc*3, Sc, 0, 0);

    int npts = Hc*PQc;
    to_global_k<<<(Bc*Nc*npts + 255)/256, 256, 0, stream>>>(qpB, Rbuf, tbuf, qgB, npts);
    to_global_k<<<(Bc*Nc*npts + 255)/256, 256, 0, stream>>>(kpB, Rbuf, tbuf, kgB, npts);
    to_global_k<<<(Bc*Nc*npts + 255)/256, 256, 0, stream>>>(vpB, Rbuf, tbuf, vgB, npts);
    sqsum_k<<<(Bc*Nc*Hc + 255)/256, 256, 0, stream>>>(qgB, qsqB);
    sqsum_k<<<(Bc*Nc*Hc + 255)/256, 256, 0, stream>>>(kgB, ksqB);

    logits_k<<<Bc*Hc*Nc, 256, 0, stream>>>(qB, kB, qgB, kgB, qsqB, ksqB, zb, pw, attnB);
    softmax_rows_k<<<Bc*Hc*Nc, 256, 0, stream>>>(attnB);

    attn_os_k<<<((long)Bc*Nc*Hc*DVc + 255)/256, 256, 0, stream>>>(attnB, vB, ocB);
    attn_og_k<<<((long)Bc*Nc*Hc*PVc*3 + 255)/256, 256, 0, stream>>>(attnB, vgB, ogB);
    finalize_pts_k<<<((long)Bc*Nc*Hc*PVc + 255)/256, 256, 0, stream>>>(ogB, Rbuf, tbuf, ocB);

    // o_z: per (b,i) GEMM  [H x N] @ [N x Z], batch = B*N, B-operand is f16 z
    gemm_launch(stream,
                attnB, 0, Nc, (long)Hc*Nc*Nc, Nc, Nc*Nc,
                z16, 1, (long)Nc*Zc, 0, Zc, Zc,
                nullptr, nullptr,
                ocB + Hc*DVc, 1, OCc, 0, Zc,
                Hc, Zc, Nc, Bc*Nc, 0, 0);

    gemm_plain(stream, ocB, OCc, Pwo, bo, sBuf, t0, Sc, Bc*Nc, Sc, OCc, 0, 0);
    ln_rows_k<<<Bc*Nc, 128, 0, stream>>>(t0, nullptr, ln_attn_g, ln_attn_b, Sc, t1, nullptr);
    gemm_plain(stream, t1, Sc, Pff1, ff_b1, nullptr, t2, Sc, Bc*Nc, Sc, Sc, 0, 1);
    gemm_plain(stream, t2, Sc, Pff2, ff_b2, nullptr, t0, Sc, Bc*Nc, Sc, Sc, 0, 1);
    gemm_plain(stream, t0, Sc, Pff3, ff_b3, t1, t2, Sc, Bc*Nc, Sc, Sc, 0, 0);
    ln_rows_k<<<Bc*Nc, 128, 0, stream>>>(t2, nullptr, ln_ff_g, ln_ff_b, Sc, t0, nullptr);
    ln_rows_k<<<Bc*Nc, 128, 0, stream>>>(t0, sBuf, ln_ipa_g, ln_ipa_b, Sc, sBuf, nullptr);
    gemm_plain(stream, sBuf, Sc, Ptw1, tb1, nullptr, t1, Sc, Bc*Nc, Sc, Sc, 0, 1);
    gemm_plain(stream, t1,   Sc, Ptw2, tb2, nullptr, t2, Sc, Bc*Nc, Sc, Sc, 0, 1);
    gemm_plain(stream, t2,   Sc, Ptw3, tb3, sBuf, t0, Sc, Bc*Nc, Sc, Sc, 0, 0);
    ln_rows_k<<<Bc*Nc, 128, 0, stream>>>(t0, nullptr, ln_t_g, ln_t_b, Sc, sBuf, nullptr);

    gemm_launch(stream, sBuf + (long)MHCc*Sc, 0, 1, (long)Nc*Sc, 0, Sc,
                Pbb.p, 1, 0, 0, Pbb.ld, Pbb.ld, bb_b, nullptr,
                updB, 1, (long)PEPc*6, 0, 6, PEPc, 6, Sc, Bc, 0, 0);
    quat_update_k<<<(Bc*PEPc + 255)/256, 256, 0, stream>>>(updB, pepR, pepT);

    gemm_launch(stream, sBuf + (long)MHCc*Sc, 0, 1, (long)Nc*Sc, 0, Sc,
                ParIn.p, 1, 0, 0, ParIn.ld, ParIn.ld, ar_in_b, aInit,
                xAr, 1, (long)PEPc*RDc, 0, RDc, PEPc, RDc, Sc, Bc, 1, 0);
    for (int r = 0; r < NRc; ++r) {
      gemm_plain(stream, xAr, RDc, Par[r][0], arw[r][1], nullptr, hAr, RDc, Bc*PEPc, RDc, RDc, 1, 0);
      gemm_plain(stream, hAr, RDc, Par[r][1], arw[r][3], xAr, xAr, RDc, Bc*PEPc, RDc, RDc, 1, 0);
    }
    gemm_plain(stream, xAr, RDc, ParO, ar_out_b, nullptr, oAr, NAc*2, Bc*PEPc, NAc*2, RDc, 1, 0);
    angles_out_k<<<(Bc*PEPc*NAc + 255)/256, 256, 0, stream>>>(oAr, dout);
  }

  // ---- pack outputs ----
  hipMemcpyAsync(dout, sBuf, (size_t)Bc*Nc*Sc*4, hipMemcpyDeviceToDevice, stream);
  pack_tail_k<<<(Bc*PEPc*9 + 255)/256, 256, 0, stream>>>(pepR, pepT, dout);
}
